// RevGAT_56856777064671
// MI455X (gfx1250) — matrix operand order repaired and checked by reading, not compiled, not run on hardware
//
#include <hip/hip_runtime.h>
#include <hip/hip_bf16.h>

#define NN 50000
#define NE 800000
#define KK 256     // IN_FEATS
#define OO 256     // 128 (W_fc) | 128 (W_res) fused output
#define NEG_SLOPE 0.2f

typedef __attribute__((ext_vector_type(16))) __bf16   v16bf;
typedef __attribute__((ext_vector_type(8)))  __bf16   v8bf;
typedef __attribute__((ext_vector_type(8)))  float    v8f;
typedef __attribute__((ext_vector_type(4)))  unsigned v4u;
typedef __attribute__((ext_vector_type(8)))  int      v8i;
typedef __attribute__((ext_vector_type(4)))  int      v4i;

// ---------- helpers ----------
__device__ __forceinline__ float leaky(float x) {
    return x > 0.0f ? x : NEG_SLOPE * x;
}
// order-preserving map f32 -> u32 so unsigned atomicMax == float max
__device__ __forceinline__ unsigned f32Key(float f) {
    unsigned u = __float_as_uint(f);
    return (u & 0x80000000u) ? ~u : (u | 0x80000000u);
}
__device__ __forceinline__ float f32FromKey(unsigned k) {
    return (k & 0x80000000u) ? __uint_as_float(k & 0x7fffffffu)
                             : __uint_as_float(~k);
}

// ---------- 0: init scratch + output ----------
__global__ void gat_init(float* outdeg, float* indeg, unsigned* emaxu,
                         float* denom, float* out) {
    int i = blockIdx.x * blockDim.x + threadIdx.x;
    if (i < NN)       { outdeg[i] = 0.0f; indeg[i] = 0.0f; }
    if (i < NN * 4)   { emaxu[i] = 0x007FFFFFu; /* key(-inf) */ denom[i] = 0.0f; }
    if (i < NN * 128) { out[i] = 0.0f; }
}

// ---------- 1: degrees ----------
__global__ void gat_degrees(const int* __restrict__ src, const int* __restrict__ dst,
                            float* outdeg, float* indeg) {
    int e = blockIdx.x * blockDim.x + threadIdx.x;
    if (e >= NE) return;
    atomicAdd(&outdeg[src[e]], 1.0f);
    atomicAdd(&indeg[dst[e]], 1.0f);
}

// ---------- 2: fused projection GEMM ----------
// P[n, 0:128]   = feat @ W_fc.T   (later overwritten in-place by feat_src)
// P[n, 128:256] = feat @ W_res.T
// One 256-thread block per 16-row stripe. A tile (16x256 f32, 16 KB) is DMA'd
// into LDS by the Tensor Data Mover once, split into bf16 hi/lo once
// cooperatively, then 8 waves each produce two 16x16 output tiles with
// bf16x3 split-precision WMMA (6 v_wmma per 32-wide K step).
__global__ __launch_bounds__(256) void gat_gemm_wmma(
    const float* __restrict__ feat, const float* __restrict__ Wfc,
    const float* __restrict__ Wres, float* __restrict__ P) {
    __shared__ __align__(16) float  Af32[16 * 256];
    __shared__ __align__(16) __bf16 Ahi[16 * 256];
    __shared__ __align__(16) __bf16 Alo[16 * 256];

    const int wave = threadIdx.x >> 5;
    const int lane = threadIdx.x & 31;
    const int rowBase = blockIdx.x * 16;   // NN/16 = 3125 blocks, exact

#if __has_builtin(__builtin_amdgcn_tensor_load_to_lds)
    if (wave == 0) {
        // ---- Tensor DMA descriptor (D#), per CDNA5 ISA 8.3/8.4 ----
        unsigned long long ga =
            (unsigned long long)(size_t)(feat + (size_t)rowBase * KK);
        unsigned ldsa = (unsigned)(unsigned long long)(size_t)(&Af32[0]);
        v4u g0 = {
            1u,                                        // count=1, user desc
            ldsa,                                      // lds_addr (bytes)
            (unsigned)(ga & 0xFFFFFFFFull),            // global_addr[31:0]
            (unsigned)((ga >> 32) & 0x01FFFFFFull)     // global_addr[56:32]
                | (2u << 30)                           // type=2 ("image")
        };
        v8i g1 = {
            0x00020000,          // workgroup_mask=0 | data_size=2 (4B)
            (int)(256u << 16),   // tensor_dim0 = 256 (lo16)
            (int)(16u  << 16),   // tensor_dim0 hi=0 | tensor_dim1 = 16 (lo16)
            (int)(256u << 16),   // tensor_dim1 hi=0 | tile_dim0 = 256
            16,                  // tile_dim1 = 16 | tile_dim2 = 0
            256,                 // tensor_dim0_stride = 256 (lo32)
            0,                   // stride hi | tensor_dim1_stride lo
            0                    // tensor_dim1_stride hi
        };
        v4i z4 = {0, 0, 0, 0};
        v8i z8 = {0, 0, 0, 0, 0, 0, 0, 0};
        __builtin_amdgcn_tensor_load_to_lds(g0, g1, z4, z4, z8, 0);
        __builtin_amdgcn_s_wait_tensorcnt(0);
    }
#else
    for (int i = threadIdx.x; i < 16 * 64; i += 256)
        ((float4*)Af32)[i] = ((const float4*)(feat + (size_t)rowBase * KK))[i];
#endif
    __syncthreads();

    // f32 -> bf16 hi/lo split, once per tile (instead of once per wave)
    for (int i = threadIdx.x; i < 16 * 256; i += 256) {
        float x = Af32[i];
        __bf16 h = (__bf16)x;
        Ahi[i] = h;
        Alo[i] = (__bf16)(x - (float)h);
    }
    __syncthreads();

    const int m = lane & 15;   // M (A) / N (B,C) index for this lane
    const int g = lane >> 4;   // lane-half selector
    const int o = wave * 16 + m;                 // 0..127
    const float* B0 = Wfc  + (size_t)o * KK;     // out col o       (W_fc)
    const float* B1 = Wres + (size_t)o * KK;     // out col o + 128 (W_res)
    const __bf16* Ah = Ahi + m * 256;
    const __bf16* Al = Alo + m * 256;

    v8f acc0 = {}, acc1 = {};
    for (int k0 = 0; k0 < KK; k0 += 32) {
        // A fragment 16x32 (MxK): lane-half g selects K sub-ranges
        // {g*8..g*8+7} and {16+g*8..16+g*8+7}: two contiguous b128 LDS loads.
        v8bf ah0 = *(const v8bf*)(Ah + k0 + g * 8);
        v8bf ah1 = *(const v8bf*)(Ah + k0 + 16 + g * 8);
        v8bf al0 = *(const v8bf*)(Al + k0 + g * 8);
        v8bf al1 = *(const v8bf*)(Al + k0 + 16 + g * 8);
        v16bf ahi = __builtin_shufflevector(ah0, ah1,
            0,1,2,3,4,5,6,7,8,9,10,11,12,13,14,15);
        v16bf alo = __builtin_shufflevector(al0, al1,
            0,1,2,3,4,5,6,7,8,9,10,11,12,13,14,15);

        // B fragments 32x16 (KxN): lane holds column o; K = g*8 + j (+8 if j >= 8)
        v16bf b0hi, b0lo, b1hi, b1lo;
#pragma unroll
        for (int j = 0; j < 16; ++j) {
            float x0 = B0[k0 + g * 8 + j + (j >= 8 ? 8 : 0)];
            float x1 = B1[k0 + g * 8 + j + (j >= 8 ? 8 : 0)];
            __bf16 h0 = (__bf16)x0, h1 = (__bf16)x1;
            b0hi[j] = h0; b0lo[j] = (__bf16)(x0 - (float)h0);
            b1hi[j] = h1; b1lo[j] = (__bf16)(x1 - (float)h1);
        }
        // bf16x3: hi*hi + hi*lo + lo*hi into f32 accumulators
        acc0 = __builtin_amdgcn_wmma_f32_16x16x32_bf16(false, ahi, false, b0hi,
                                                       (short)0, acc0, false, false);
        acc0 = __builtin_amdgcn_wmma_f32_16x16x32_bf16(false, ahi, false, b0lo,
                                                       (short)0, acc0, false, false);
        acc0 = __builtin_amdgcn_wmma_f32_16x16x32_bf16(false, alo, false, b0hi,
                                                       (short)0, acc0, false, false);
        acc1 = __builtin_amdgcn_wmma_f32_16x16x32_bf16(false, ahi, false, b1hi,
                                                       (short)0, acc1, false, false);
        acc1 = __builtin_amdgcn_wmma_f32_16x16x32_bf16(false, ahi, false, b1lo,
                                                       (short)0, acc1, false, false);
        acc1 = __builtin_amdgcn_wmma_f32_16x16x32_bf16(false, alo, false, b1hi,
                                                       (short)0, acc1, false, false);
    }
    // C/D layout: vgpr r -> row M = r + 8*g, col = lane&15
#pragma unroll
    for (int r = 0; r < 8; ++r) {
        int row = rowBase + r + 8 * g;
        P[(size_t)row * OO + o]       = acc0[r];
        P[(size_t)row * OO + 128 + o] = acc1[r];
    }
}

// ---------- 3: per-node prep: norm, el/er, scale h -> feat_src in place ----------
__global__ __launch_bounds__(256) void gat_node_prep(
    float* __restrict__ P, const float* __restrict__ outdeg,
    const float* __restrict__ attn_l, const float* __restrict__ attn_r,
    float* __restrict__ el, float* __restrict__ er) {
    const int wave = threadIdx.x >> 5;
    const int lane = threadIdx.x & 31;
    const int n = blockIdx.x * 8 + wave;
    if (n >= NN) return;
    const float norm = rsqrtf(fmaxf(outdeg[n], 1.0f));
    float elp[4], erp[4];
#pragma unroll
    for (int h = 0; h < 4; ++h) {
        const int idx = h * 32 + lane;
        float hv = P[(size_t)n * OO + idx];   // pre-norm projection
        float fs = hv * norm;                 // feat_src
        P[(size_t)n * OO + idx] = fs;         // overwrite in place
        elp[h] = fs * attn_l[idx];
        erp[h] = hv * attn_r[idx];            // er uses pre-norm h
    }
#pragma unroll
    for (int off = 16; off > 0; off >>= 1) {
#pragma unroll
        for (int h = 0; h < 4; ++h) {
            elp[h] += __shfl_xor(elp[h], off, 32);
            erp[h] += __shfl_xor(erp[h], off, 32);
        }
    }
    if (lane == 0) {
#pragma unroll
        for (int h = 0; h < 4; ++h) {
            el[n * 4 + h] = elp[h];
            er[n * 4 + h] = erp[h];
        }
    }
}

// ---------- 4: edge pass A — segment max over dst ----------
__global__ void gat_edge_max(const int* __restrict__ src, const int* __restrict__ dst,
                             const float* __restrict__ el, const float* __restrict__ er,
                             unsigned* __restrict__ emaxu) {
    int e = blockIdx.x * blockDim.x + threadIdx.x;
    if (e >= NE) return;
    const int s = src[e], d = dst[e];
    const float4 L = *(const float4*)(el + 4 * (size_t)s);
    const float4 R = *(const float4*)(er + 4 * (size_t)d);
    float sc[4] = {L.x + R.x, L.y + R.y, L.z + R.z, L.w + R.w};
#pragma unroll
    for (int h = 0; h < 4; ++h)
        atomicMax(&emaxu[(size_t)d * 4 + h], f32Key(leaky(sc[h])));
}

// ---------- 5: decode emax keys, guard empty segments like the reference ----------
__global__ void gat_emax_fin(const unsigned* __restrict__ emaxu,
                             float* __restrict__ emaxf) {
    int i = blockIdx.x * blockDim.x + threadIdx.x;
    if (i >= NN * 4) return;
    float m = f32FromKey(emaxu[i]);
    bool finite = (m == m) && (fabsf(m) < __builtin_inff());
    emaxf[i] = finite ? m : 0.0f;
}

// ---------- 6: edge pass B — exp + segment sum (denominator) ----------
__global__ void gat_edge_sum(const int* __restrict__ src, const int* __restrict__ dst,
                             const float* __restrict__ el, const float* __restrict__ er,
                             const float* __restrict__ emaxf, float* __restrict__ denom) {
    int e = blockIdx.x * blockDim.x + threadIdx.x;
    if (e >= NE) return;
    const int s = src[e], d = dst[e];
    const float4 L = *(const float4*)(el + 4 * (size_t)s);
    const float4 R = *(const float4*)(er + 4 * (size_t)d);
    const float4 M = *(const float4*)(emaxf + 4 * (size_t)d);
    float sc[4] = {L.x + R.x, L.y + R.y, L.z + R.z, L.w + R.w};
    float mx[4] = {M.x, M.y, M.z, M.w};
#pragma unroll
    for (int h = 0; h < 4; ++h)
        atomicAdd(&denom[(size_t)d * 4 + h], __expf(leaky(sc[h]) - mx[h]));
}

// ---------- 7: edge pass C — normalize + weighted scatter (wave per edge) ----------
__global__ __launch_bounds__(256) void gat_scatter(
    const int* __restrict__ src, const int* __restrict__ dst,
    const float* __restrict__ el, const float* __restrict__ er,
    const float* __restrict__ emaxf, const float* __restrict__ denom,
    const float* __restrict__ P, float* __restrict__ out) {
    const int wave = threadIdx.x >> 5;
    const int lane = threadIdx.x & 31;
    const int e = blockIdx.x * 8 + wave;
    if (e >= NE) return;
    const int s = src[e], d = dst[e];
    const float4 L  = *(const float4*)(el + 4 * (size_t)s);
    const float4 R  = *(const float4*)(er + 4 * (size_t)d);
    const float4 M  = *(const float4*)(emaxf + 4 * (size_t)d);
    const float4 Dn = *(const float4*)(denom + 4 * (size_t)d);
    float a[4];
    a[0] = __expf(leaky(L.x + R.x) - M.x) / Dn.x;
    a[1] = __expf(leaky(L.y + R.y) - M.y) / Dn.y;
    a[2] = __expf(leaky(L.z + R.z) - M.z) / Dn.z;
    a[3] = __expf(leaky(L.w + R.w) - M.w) / Dn.w;
    const float* fs = P + (size_t)s * OO;   // feat_src (cols 0..127)
    float* o = out + (size_t)d * 128;
#pragma unroll
    for (int h = 0; h < 4; ++h)
        atomicAdd(&o[h * 32 + lane], fs[h * 32 + lane] * a[h]);
}

// ---------- 8: finalize: * sqrt(in_deg) + residual ----------
__global__ void gat_finalize(float* __restrict__ out, const float* __restrict__ P,
                             const float* __restrict__ indeg) {
    int i = blockIdx.x * blockDim.x + threadIdx.x;
    if (i >= NN * 128) return;
    const int n = i >> 7, c = i & 127;
    const float sc = sqrtf(fmaxf(indeg[n], 1.0f));
    out[i] = out[i] * sc + P[(size_t)n * OO + 128 + c];
}

extern "C" void kernel_launch(void* const* d_in, const int* in_sizes, int n_in,
                              void* d_out, int out_size, void* d_ws, size_t ws_size,
                              hipStream_t stream) {
    const float* feat   = (const float*)d_in[0];
    const int*   src    = (const int*)d_in[1];
    const int*   dst    = (const int*)d_in[2];
    const float* W_fc   = (const float*)d_in[3];
    const float* attn_l = (const float*)d_in[4];
    const float* attn_r = (const float*)d_in[5];
    const float* W_res  = (const float*)d_in[6];
    float* out = (float*)d_out;

    // workspace layout (floats): ~56 MB
    float*    P      = (float*)d_ws;            // NN*256  [h | res]
    float*    el     = P + (size_t)NN * 256;    // NN*4
    float*    er     = el + (size_t)NN * 4;     // NN*4
    float*    emaxf  = er + (size_t)NN * 4;     // NN*4
    unsigned* emaxu  = (unsigned*)(emaxf + (size_t)NN * 4);   // NN*4
    float*    denom  = (float*)(emaxu + (size_t)NN * 4);      // NN*4
    float*    outdeg = denom + (size_t)NN * 4;  // NN
    float*    indeg  = outdeg + (size_t)NN;     // NN

    const int TB = 256;
    gat_init<<<(NN * 128 + TB - 1) / TB, TB, 0, stream>>>(outdeg, indeg, emaxu, denom, out);
    gat_degrees<<<(NE + TB - 1) / TB, TB, 0, stream>>>(src, dst, outdeg, indeg);
    gat_gemm_wmma<<<NN / 16, 256, 0, stream>>>(feat, W_fc, W_res, P);
    gat_node_prep<<<NN / 8, 256, 0, stream>>>(P, outdeg, attn_l, attn_r, el, er);
    gat_edge_max<<<(NE + TB - 1) / TB, TB, 0, stream>>>(src, dst, el, er, emaxu);
    gat_emax_fin<<<(NN * 4 + TB - 1) / TB, TB, 0, stream>>>(emaxu, emaxf);
    gat_edge_sum<<<(NE + TB - 1) / TB, TB, 0, stream>>>(src, dst, el, er, emaxf, denom);
    gat_scatter<<<(NE + 7) / 8, 256, 0, stream>>>(src, dst, el, er, emaxf, denom, P, out);
    gat_finalize<<<(NN * 128 + TB - 1) / TB, TB, 0, stream>>>(out, P, indeg);
}
